// BidirectionalMamba_77506979823809
// MI455X (gfx1250) — matrix operand order repaired and checked by reading, not compiled, not run on hardware
//
#include <hip/hip_runtime.h>

// ---------------------------------------------------------------------------
// Bidirectional Mamba for MI455X (gfx1250, wave32, WMMA).
// GEMMs: v_wmma_f32_16x16x32_bf16 with double-buffered async copies
// (global_load_async_to_lds_b128 / s_wait_asynccnt) staging A+B tiles in LDS.
// Scan: lane-per-(b,e,n) + shfl_xor reduce + global_prefetch_b8.
// ---------------------------------------------------------------------------

typedef __attribute__((ext_vector_type(16))) __bf16 v16bf;
typedef __attribute__((ext_vector_type(8)))  __bf16 v8bf;
typedef __attribute__((ext_vector_type(8)))  float  v8f;

#define D_MODEL 1024
#define E_DIM   2048
#define NSTATE  16
#define DCONV   4
#define DT_RANK 64
#define BATCH   4
#define SEQ     2048
#define NROWS   (BATCH * SEQ)            /* 8192  */
#define XZ_COLS (2 * E_DIM)              /* 4096  */
#define PROJ_PAD 128                     /* 96 cols padded to 128 */

// GEMM tiling: block = 4 waves (2m x 2n), block tile 64(M) x 128(N), K step 32.
// LDS staging: A tile 64 rows x 64B (padded to 80B), B tile 128 cols x 64B
// (padded to 80B). 80B = 5*16B keeps ds_load_b128 alignment and spreads the
// 16 fragment lanes over 16 distinct banks (stride 20 banks, cycle 16).
#define ROW_B    80
#define A_BYTES  (64 * ROW_B)            /* 5120  */
#define B_BYTES  (128 * ROW_B)           /* 10240 */
#define BUF_B    (A_BYTES + B_BYTES)     /* 15360 */

static __device__ __forceinline__ __bf16 f2bf(float f) { return (__bf16)f; }
static __device__ __forceinline__ float silu_f(float x) { return x / (1.f + __expf(-x)); }

static __device__ __forceinline__ void async_cp16(unsigned lds_addr, const void* gaddr)
{
    // 16B global -> LDS, tracked by ASYNCcnt (ISA 15.18.3 op 98).
    asm volatile("global_load_async_to_lds_b128 %0, %1, off"
                 :: "v"(lds_addr), "v"((unsigned long long)(uintptr_t)gaddr)
                 : "memory");
}

// ---------------------------------------------------------------------------
// Weight prep: dst[n*K + k] = bf16(src[k*N + n]), zero-padded to Npad rows.
// ---------------------------------------------------------------------------
__global__ __launch_bounds__(256)
void transpose_bf16_kernel(const float* __restrict__ src, int K, int N, int Npad,
                           __bf16* __restrict__ dst)
{
    long idx = (long)blockIdx.x * blockDim.x + threadIdx.x;
    if (idx >= (long)Npad * K) return;
    int k = (int)(idx % K);
    int n = (int)(idx / K);
    dst[idx] = (n < N) ? f2bf(src[(long)k * N + n]) : (__bf16)0.f;
}

__global__ __launch_bounds__(256)
void cvt_bf16_kernel(const float* __restrict__ src, __bf16* __restrict__ dst, long n)
{
    long idx = (long)blockIdx.x * blockDim.x + threadIdx.x;
    if (idx < n) dst[idx] = f2bf(src[idx]);
}

// ---------------------------------------------------------------------------
// WMMA GEMM: C[M,N] (+)= A[M,K] @ B[K,N].
//  A : bf16 row-major (lda); B : bf16 pre-transposed BT[N][K].
//  M%64==0, N%128==0, K%32==0 (no guards anywhere).
//  Stages: each wave issues 2 A-chunk + 4 B-chunk async copies per k-step
//  (16B per lane), double buffered; s_wait_asynccnt 0x6 + barrier before use.
//  Fragments from LDS per ISA 7.12.2 layouts:
//   A 16x32 bf16: lane half=lane>>4, M=lane&15; bytes [half*16,+15] & +32.
//   B 32x16 bf16: N=lane&15, bytes [half*16,+15] & +32.
//   C/D f32: VGPR v -> M = v + half*8, N = lane&15.
//  arow_flip / crow_flip: map row (b,l) <-> (b, SEQ-1-l) for the bwd pass.
//  epi: 0 none | 1 bias+softplus | 2 also store bf16 shadow Cbf.
// ---------------------------------------------------------------------------
__global__ __launch_bounds__(128)
void gemm_bf16_wmma(const __bf16* __restrict__ A, int lda, int arow_flip,
                    const __bf16* __restrict__ BT, int K,
                    float* __restrict__ C, int ldc, int crow_flip, int accum,
                    int epi, const float* __restrict__ bias,
                    __bf16* __restrict__ Cbf)
{
    __shared__ char lds_raw[2 * BUF_B];

    const int lane   = threadIdx.x & 31;
    const int wave   = threadIdx.x >> 5;
    const int wave_m = wave >> 1;
    const int wave_n = wave & 1;
    const int half   = lane >> 4;
    const int lm     = lane & 15;
    const unsigned lds0 = (unsigned)(uintptr_t)lds_raw;

    // --- per-lane staging descriptors (global ptr at k=0, LDS offset) ---
    const __bf16* a_gp[2];   unsigned a_lo[2];
#pragma unroll
    for (int i = 0; i < 2; ++i) {
        int chunk = (wave * 2 + i) * 32 + lane;   // 0..255
        int r = chunk >> 2, c = chunk & 3;        // r: tile row, c: 16B chunk
        int row  = blockIdx.y * 64 + r;
        int srow = row;
        if (arow_flip) { int b = row / SEQ, l = row % SEQ; srow = b * SEQ + (SEQ - 1 - l); }
        a_gp[i] = A + (long)srow * lda + c * 8;
        a_lo[i] = r * ROW_B + c * 16;
    }
    const __bf16* b_gp[4];   unsigned b_lo[4];
#pragma unroll
    for (int i = 0; i < 4; ++i) {
        int chunk = (wave * 4 + i) * 32 + lane;   // 0..511
        int col = chunk >> 2, c = chunk & 3;      // col: tile col (N)
        b_gp[i] = BT + (long)(blockIdx.x * 128 + col) * K + c * 8;
        b_lo[i] = A_BYTES + col * ROW_B + c * 16;
    }

    auto stage = [&](int buf, int k0) {
        const unsigned base = lds0 + buf * BUF_B;
#pragma unroll
        for (int i = 0; i < 2; ++i) async_cp16(base + a_lo[i], a_gp[i] + k0);
#pragma unroll
        for (int i = 0; i < 4; ++i) async_cp16(base + b_lo[i], b_gp[i] + k0);
    };

    v8f acc[2][4] = {};
    const int nk = K / 32;

    stage(0, 0);
    for (int ki = 0; ki < nk; ++ki) {
        const int buf = ki & 1;
        if (ki + 1 < nk) {
            stage(buf ^ 1, (ki + 1) * 32);
            asm volatile("s_wait_asynccnt 0x6" ::: "memory");  // stage ki landed
        } else {
            asm volatile("s_wait_asynccnt 0x0" ::: "memory");
        }
        __syncthreads();

        const char* Ab = lds_raw + buf * BUF_B;
        const char* Bb = Ab + A_BYTES;

        v16bf afrag[2];
#pragma unroll
        for (int mt = 0; mt < 2; ++mt) {
            const char* p = Ab + (wave_m * 32 + mt * 16 + lm) * ROW_B + half * 16;
            v8bf lo = *(const v8bf*)p;
            v8bf hi = *(const v8bf*)(p + 32);
            afrag[mt] = __builtin_shufflevector(lo, hi, 0, 1, 2, 3, 4, 5, 6, 7,
                                                8, 9, 10, 11, 12, 13, 14, 15);
        }
        v16bf bfrag[4];
#pragma unroll
        for (int nt = 0; nt < 4; ++nt) {
            const char* p = Bb + (wave_n * 64 + nt * 16 + lm) * ROW_B + half * 16;
            v8bf lo = *(const v8bf*)p;
            v8bf hi = *(const v8bf*)(p + 32);
            bfrag[nt] = __builtin_shufflevector(lo, hi, 0, 1, 2, 3, 4, 5, 6, 7,
                                                8, 9, 10, 11, 12, 13, 14, 15);
        }

#pragma unroll
        for (int mt = 0; mt < 2; ++mt)
#pragma unroll
            for (int nt = 0; nt < 4; ++nt)
                acc[mt][nt] = __builtin_amdgcn_wmma_f32_16x16x32_bf16(
                    false, afrag[mt], false, bfrag[nt],
                    (short)0, acc[mt][nt], false, false);

        __syncthreads();   // all waves done reading buf before it is restaged
    }

    const int m0 = blockIdx.y * 64 + wave_m * 32;
    const int n0 = blockIdx.x * 128 + wave_n * 64;
#pragma unroll
    for (int mt = 0; mt < 2; ++mt) {
#pragma unroll
        for (int nt = 0; nt < 4; ++nt) {
            const int col = n0 + nt * 16 + lm;
#pragma unroll
            for (int v = 0; v < 8; ++v) {
                int   row = m0 + mt * 16 + v + half * 8;
                float val = acc[mt][nt][v];
                if (epi == 1) {                     // dt_b bias + softplus
                    val += bias[col];
                    val = (val > 20.f) ? val : __logf(1.f + __expf(val));
                }
                int orow = row;
                if (crow_flip) { int b = row / SEQ, l = row % SEQ; orow = b * SEQ + (SEQ - 1 - l); }
                float* cp = C + (long)orow * ldc + col;
                if (accum) *cp += val; else *cp = val;
                if (epi == 2) Cbf[(long)orow * ldc + col] = f2bf(val);
            }
        }
    }
}

// ---------------------------------------------------------------------------
// Depthwise causal conv1d (width 4) + bias + SiLU; writes fp32 (scan) and
// bf16 (GEMM A-operand) copies of xi.
// ---------------------------------------------------------------------------
__global__ __launch_bounds__(256)
void conv_silu_kernel(const float* __restrict__ xz, const float* __restrict__ cw,
                      const float* __restrict__ cb, float* __restrict__ xi,
                      __bf16* __restrict__ xibf)
{
    long idx = (long)blockIdx.x * blockDim.x + threadIdx.x;
    if (idx >= (long)NROWS * E_DIM) return;
    int  e = (int)(idx % E_DIM);
    long r = idx / E_DIM;          // b*SEQ + l
    int  l = (int)(r % SEQ);
    float acc = cb[e];
#pragma unroll
    for (int j = 0; j < DCONV; ++j) {
        int ls = l - (DCONV - 1) + j;
        if (ls >= 0)
            acc += xz[(r - (DCONV - 1) + j) * XZ_COLS + e] * cw[e * DCONV + j];
    }
    float s = silu_f(acc);
    xi[idx]   = s;
    xibf[idx] = f2bf(s);
}

// ---------------------------------------------------------------------------
// Selective scan. One lane per (b, e, n): 2 channels per wave32
// (lanes 0..15 -> e0, lanes 16..31 -> e1; n = lane&15).
// h = exp(delta*A)*h + delta*u*B ; y = sum_n h*C (4-step shfl_xor reduce),
// fused + Dp*u skip and silu(z) gate; y stored bf16 for the output GEMM.
// ---------------------------------------------------------------------------
__global__ __launch_bounds__(256)
void scan_kernel(const float* __restrict__ delta, const float* __restrict__ xi,
                 const float* __restrict__ proj,  const float* __restrict__ xz,
                 const float* __restrict__ A_log, const float* __restrict__ Dp,
                 __bf16* __restrict__ ybf)
{
    const int lane = threadIdx.x & 31;
    const int wave = threadIdx.x >> 5;
    const int n    = lane & 15;
    const int b    = blockIdx.y;
    const int e    = blockIdx.x * 16 + wave * 2 + (lane >> 4);

    const float Aen = -__expf(A_log[e * NSTATE + n]);
    const float dpe = Dp[e];
    float h = 0.f;

    const long base = (long)b * SEQ;
    for (int l = 0; l < SEQ; ++l) {
        long  r  = base + l;
        float d  = delta[r * E_DIM + e];
        float u  = xi[r * E_DIM + e];
        float bn = proj[r * PROJ_PAD + DT_RANK + n];
        float cn = proj[r * PROJ_PAD + DT_RANK + NSTATE + n];

        if (l + 1 < SEQ) {   // hide next-step latency (global_prefetch_b8)
            __builtin_prefetch(&delta[(r + 1) * E_DIM + e], 0, 1);
            __builtin_prefetch(&xi[(r + 1) * E_DIM + e], 0, 1);
            __builtin_prefetch(&proj[(r + 1) * PROJ_PAD + DT_RANK + n], 0, 1);
        }

        h = __expf(d * Aen) * h + d * u * bn;

        float part = h * cn;               // reduce over 16 states in-wave
        part += __shfl_xor(part, 1, 32);
        part += __shfl_xor(part, 2, 32);
        part += __shfl_xor(part, 4, 32);
        part += __shfl_xor(part, 8, 32);

        if (n == 0) {
            float z = xz[r * XZ_COLS + E_DIM + e];
            ybf[r * E_DIM + e] = f2bf((part + dpe * u) * silu_f(z));
        }
    }
}

// ---------------------------------------------------------------------------
static inline size_t align256(size_t x) { return (x + 255) & ~(size_t)255; }

extern "C" void kernel_launch(void* const* d_in, const int* in_sizes, int n_in,
                              void* d_out, int out_size, void* d_ws, size_t ws_size,
                              hipStream_t stream)
{
    (void)in_sizes; (void)out_size;
    if (n_in < 19) return;

    const float* x = (const float*)d_in[0];
    float* out = (float*)d_out;

    // ---- workspace carve-up (all offsets 256B aligned) ----
    char*  p = (char*)d_ws;
    size_t off = 0;
    auto carve = [&](size_t bytes) { void* q = p + off; off = align256(off + bytes); return q; };

    float*  xz     = (float*) carve((size_t)NROWS * XZ_COLS * 4);   // 134 MB
    float*  xi32   = (float*) carve((size_t)NROWS * E_DIM * 4);     //  67 MB
    __bf16* xibf   = (__bf16*)carve((size_t)NROWS * E_DIM * 2);     //  33 MB
    float*  proj   = (float*) carve((size_t)NROWS * PROJ_PAD * 4);  //   4 MB
    __bf16* projbf = (__bf16*)carve((size_t)NROWS * PROJ_PAD * 2);  //   2 MB
    float*  delta  = (float*) carve((size_t)NROWS * E_DIM * 4);     //  67 MB
    __bf16* ybf    = (__bf16*)carve((size_t)NROWS * E_DIM * 2);     //  33 MB
    __bf16* xbf    = (__bf16*)carve((size_t)NROWS * D_MODEL * 2);   //  16 MB
    __bf16* in_wT  = (__bf16*)carve((size_t)XZ_COLS * D_MODEL * 2); //   8 MB
    __bf16* xprojT = (__bf16*)carve((size_t)PROJ_PAD * E_DIM * 2);
    __bf16* dt_wT  = (__bf16*)carve((size_t)E_DIM * DT_RANK * 2);
    __bf16* out_wT = (__bf16*)carve((size_t)D_MODEL * E_DIM * 2);   //   4 MB
    if (off > ws_size) return;

    // x -> bf16 once (both directions read it, bwd via row-flip mapping).
    {
        long nels = (long)NROWS * D_MODEL;
        cvt_bf16_kernel<<<(int)((nels + 255) / 256), 256, 0, stream>>>(x, xbf, nels);
    }

    for (int dir = 0; dir < 2; ++dir) {
        const float* in_w    = (const float*)d_in[1 + dir * 9 + 0];
        const float* conv_w  = (const float*)d_in[1 + dir * 9 + 1];
        const float* conv_b  = (const float*)d_in[1 + dir * 9 + 2];
        const float* xproj_w = (const float*)d_in[1 + dir * 9 + 3];
        const float* dt_w    = (const float*)d_in[1 + dir * 9 + 4];
        const float* dt_b    = (const float*)d_in[1 + dir * 9 + 5];
        const float* A_log   = (const float*)d_in[1 + dir * 9 + 6];
        const float* Dp      = (const float*)d_in[1 + dir * 9 + 7];
        const float* out_w   = (const float*)d_in[1 + dir * 9 + 8];

        // Weight staging: transpose + bf16 (contiguous-K fragment loads).
        transpose_bf16_kernel<<<(XZ_COLS * D_MODEL) / 256, 256, 0, stream>>>(
            in_w, D_MODEL, XZ_COLS, XZ_COLS, in_wT);
        transpose_bf16_kernel<<<(PROJ_PAD * E_DIM) / 256, 256, 0, stream>>>(
            xproj_w, E_DIM, 96, PROJ_PAD, xprojT);
        transpose_bf16_kernel<<<(E_DIM * DT_RANK) / 256, 256, 0, stream>>>(
            dt_w, DT_RANK, E_DIM, E_DIM, dt_wT);
        transpose_bf16_kernel<<<(D_MODEL * E_DIM) / 256, 256, 0, stream>>>(
            out_w, E_DIM, D_MODEL, D_MODEL, out_wT);

        // 1) xz = x[dir] @ in_w            (8192 x 4096, K=1024)
        gemm_bf16_wmma<<<dim3(XZ_COLS / 128, NROWS / 64), 128, 0, stream>>>(
            xbf, D_MODEL, dir, in_wT, D_MODEL,
            xz, XZ_COLS, 0, 0, 0, nullptr, nullptr);

        // 2) xi = silu(causal_conv(xz[:, :E]) + conv_b)  (fp32 + bf16)
        conv_silu_kernel<<<(NROWS * E_DIM) / 256, 256, 0, stream>>>(
            xz, conv_w, conv_b, xi32, xibf);

        // 3) proj = xi @ xproj_w           (8192 x 128pad, K=2048) + bf16 shadow
        gemm_bf16_wmma<<<dim3(1, NROWS / 64), 128, 0, stream>>>(
            xibf, E_DIM, 0, xprojT, E_DIM,
            proj, PROJ_PAD, 0, 0, 2, nullptr, projbf);

        // 4) delta = softplus(proj[:, :64] @ dt_w + dt_b) (8192 x 2048, K=64)
        gemm_bf16_wmma<<<dim3(E_DIM / 128, NROWS / 64), 128, 0, stream>>>(
            projbf, PROJ_PAD, 0, dt_wT, DT_RANK,
            delta, E_DIM, 0, 0, 1, dt_b, nullptr);

        // 5) selective scan + D skip + silu(z) gate -> ybf
        scan_kernel<<<dim3(E_DIM / 16, BATCH), 256, 0, stream>>>(
            delta, xi32, proj, xz, A_log, Dp, ybf);

        // 6) out (+)= y @ out_w  (bwd: row-flipped accumulate onto fwd)
        gemm_bf16_wmma<<<dim3(D_MODEL / 128, NROWS / 64), 128, 0, stream>>>(
            ybf, E_DIM, 0, out_wT, E_DIM,
            out, D_MODEL, dir, dir, 0, nullptr, nullptr);
    }
}
